// CGC3_48163763257633
// MI455X (gfx1250) — matrix operand order repaired and checked by reading, not compiled, not run on hardware
//
#include <hip/hip_runtime.h>
#include <hip/hip_bf16.h>

// ---------------------------------------------------------------------------
// CGC3 mixture-of-experts for MI455X (gfx1250, wave32, WMMA).
// Grouped GEMMs run on v_wmma_f32_16x16x32_bf16 (compute-bound workload;
// bf16 inputs + f32 accumulation). 128x128 block tile, double-buffered LDS,
// v_perm_b32 register-transposed B staging, global prefetch.
// ---------------------------------------------------------------------------

typedef __attribute__((ext_vector_type(16))) __bf16 v16bf;
typedef __attribute__((ext_vector_type(8)))  float  v8f;

#define B_SZ 8192
#define D_SZ 1024
#define H_SZ 1024
#define O_SZ 512
#define E_SZ 8
#define NGATE 20   // 6 (task0) + 6 (task1) + 8 (shared)

__device__ __forceinline__ unsigned short f32_to_bf16_rne(float f) {
    unsigned int u = __float_as_uint(f);
    unsigned int r = u + 0x7FFFu + ((u >> 16) & 1u);   // round-to-nearest-even
    return (unsigned short)(r >> 16);
}

// ------------------------- f32 -> bf16 conversion ---------------------------
__global__ __launch_bounds__(256)
void f2bf_kernel(const float* __restrict__ src, unsigned short* __restrict__ dst,
                 long long n) {
    long long i = (long long)blockIdx.x * blockDim.x + threadIdx.x;
    long long stride = (long long)gridDim.x * blockDim.x;
    for (; i < n; i += stride) dst[i] = f32_to_bf16_rne(src[i]);
}

// ------------------------- grouped GEMM + bias + relu -----------------------
// Block tile 128(M) x 128(N), K-step 32, double-buffered LDS.
// 8 waves in 4(M) x 2(N); each wave owns 32x64 = 2x4 wmma_f32_16x16x32_bf16
// fragments => 8 WMMAs per wave per K-step.
// A: [M][K] bf16 row-major (per expert via strideAe; 0 => shared input x)
// W: [K][N] bf16 row-major per expert; staged into LDS transposed [n][k]
//    via a 4x4 v_perm_b32 register transpose (k-contiguous => ds_store_b64).
// out: bf16 (layer1 -> h) or f32 (layer2 -> eo), bias + relu fused.
__global__ __launch_bounds__(256)
void gemm_bias_relu(const unsigned short* __restrict__ Abase, long long strideAe,
                    const unsigned short* __restrict__ Wbase, long long strideWe,
                    const float* __restrict__ biasBase,      long long strideBe,
                    unsigned short* __restrict__ outB, float* __restrict__ outF,
                    long long strideOe, int M, int N, int K)
{
    __shared__ unsigned short lds_a[2][128 * 32];   // [m][k]   2 x 8 KB
    __shared__ unsigned short lds_b[2][128 * 32];   // [n][k]   2 x 8 KB (transposed)

    const int e = blockIdx.z;
    const unsigned short* A = Abase + (long long)e * strideAe;
    const unsigned short* W = Wbase + (long long)e * strideWe;
    const float* bias = biasBase + (long long)e * strideBe;

    const int tid   = threadIdx.x;
    const int lane  = tid & 31;
    const int wave  = tid >> 5;
    const int waveM = wave & 3;     // 4 waves along M (32 rows each)
    const int waveN = wave >> 2;    // 2 waves along N (64 cols each)

    const int tileM = blockIdx.y * 128;
    const int tileN = blockIdx.x * 128;

    v8f acc[2][4];
    #pragma unroll
    for (int mi = 0; mi < 2; ++mi)
        #pragma unroll
        for (int ni = 0; ni < 4; ++ni)
            acc[mi][ni] = (v8f){0.f,0.f,0.f,0.f,0.f,0.f,0.f,0.f};

    // A-tile map: thread loads 16 contiguous bf16 (two 16 B chunks).
    const int a_row  = tid >> 1;          // 0..127
    const int a_koff = (tid & 1) * 16;    // 0 or 16
    // B-tile map: thread owns a 4(k) x 4(n) micro-block.
    const int b_k0 = (tid >> 5) * 4;      // 0..28
    const int b_n0 = (tid & 31) * 4;      // 0..124

    const int lrow = lane & 15;
    const int hi   = lane >> 4;           // lane half (ISA fragment layout)

    uint4 areg0, areg1;
    uint2 breg[4];                        // breg[r] = row (k0+r), 4 bf16 cols n0..n0+3

    auto globalLoad = [&](int kt) {
        const uint4* ag = (const uint4*)(A + (long long)(tileM + a_row) * K
                                           + (kt * 32 + a_koff));
        areg0 = ag[0];
        areg1 = ag[1];
        #pragma unroll
        for (int r = 0; r < 4; ++r)
            breg[r] = *(const uint2*)(W + (long long)(kt * 32 + b_k0 + r) * N
                                        + tileN + b_n0);
    };
    auto ldsStore = [&](int buf) {
        *(uint4*)&lds_a[buf][a_row * 32 + a_koff]     = areg0;
        *(uint4*)&lds_a[buf][a_row * 32 + a_koff + 8] = areg1;
        // 4x4 ushort transpose: each output word = {rowA.u16, rowB.u16} = 1 v_perm_b32
        const unsigned SEL_LO = 0x05040100u;  // low  ushort of src1 | low  ushort of src0
        const unsigned SEL_HI = 0x07060302u;  // high ushort of src1 | high ushort of src0
        uint2 c0, c1, c2, c3;
        c0.x = __builtin_amdgcn_perm(breg[1].x, breg[0].x, SEL_LO);
        c0.y = __builtin_amdgcn_perm(breg[3].x, breg[2].x, SEL_LO);
        c1.x = __builtin_amdgcn_perm(breg[1].x, breg[0].x, SEL_HI);
        c1.y = __builtin_amdgcn_perm(breg[3].x, breg[2].x, SEL_HI);
        c2.x = __builtin_amdgcn_perm(breg[1].y, breg[0].y, SEL_LO);
        c2.y = __builtin_amdgcn_perm(breg[3].y, breg[2].y, SEL_LO);
        c3.x = __builtin_amdgcn_perm(breg[1].y, breg[0].y, SEL_HI);
        c3.y = __builtin_amdgcn_perm(breg[3].y, breg[2].y, SEL_HI);
        *(uint2*)&lds_b[buf][(b_n0 + 0) * 32 + b_k0] = c0;   // k-contiguous stores
        *(uint2*)&lds_b[buf][(b_n0 + 1) * 32 + b_k0] = c1;
        *(uint2*)&lds_b[buf][(b_n0 + 2) * 32 + b_k0] = c2;
        *(uint2*)&lds_b[buf][(b_n0 + 3) * 32 + b_k0] = c3;
    };

    const int kTiles = K >> 5;
    globalLoad(0);
    ldsStore(0);
    __syncthreads();

    for (int kt = 0; kt < kTiles; ++kt) {
        const int buf = kt & 1;
        if (kt + 1 < kTiles) globalLoad(kt + 1);
        if (kt + 2 < kTiles) {   // warm L2/WGP$ for the tile after next
            __builtin_prefetch(A + (long long)(tileM + a_row) * K
                                 + ((kt + 2) * 32 + a_koff), 0, 1);
            __builtin_prefetch(W + (long long)((kt + 2) * 32 + b_k0) * N
                                 + tileN + b_n0, 0, 1);
        }

        // ---- fragments (ISA 16-bit A 16x32 / B 32x16 wave32 layouts) ----
        union Frag { uint4 q[2]; v16bf v; };
        Frag fa[2], fb[4];
        #pragma unroll
        for (int mi = 0; mi < 2; ++mi) {
            int row = waveM * 32 + mi * 16 + lrow;
            int k0  = hi * 8;  // lanes 0-15: K 0..7 & 16..23; lanes 16-31: 8..15 & 24..31
            fa[mi].q[0] = *(const uint4*)&lds_a[buf][row * 32 + k0];
            fa[mi].q[1] = *(const uint4*)&lds_a[buf][row * 32 + k0 + 16];
        }
        #pragma unroll
        for (int ni = 0; ni < 4; ++ni) {
            int col = waveN * 64 + ni * 16 + lrow;
            int kb  = hi * 8;  // lanes 0-15: K 0..7 & 16..23; lanes 16-31: 8..15 & 24..31
            fb[ni].q[0] = *(const uint4*)&lds_b[buf][col * 32 + kb];
            fb[ni].q[1] = *(const uint4*)&lds_b[buf][col * 32 + kb + 16];
        }

        #pragma unroll
        for (int mi = 0; mi < 2; ++mi)
            #pragma unroll
            for (int ni = 0; ni < 4; ++ni)
                acc[mi][ni] = __builtin_amdgcn_wmma_f32_16x16x32_bf16(
                    false, fa[mi].v, false, fb[ni].v,
                    (short)0, acc[mi][ni], false, false);

        if (kt + 1 < kTiles) {
            __syncthreads();          // everyone done reading buf^1
            ldsStore(buf ^ 1);
            __syncthreads();          // buf^1 ready for next iteration
        }
    }

    // ---- epilogue: bias + relu; C/D layout: VGPR r -> M=r (lanes 0-15) / M=r+8 ----
    const int rofs = hi * 8;
    #pragma unroll
    for (int mi = 0; mi < 2; ++mi) {
        #pragma unroll
        for (int ni = 0; ni < 4; ++ni) {
            int col = tileN + waveN * 64 + ni * 16 + lrow;
            float bv = bias[col];
            #pragma unroll
            for (int r = 0; r < 8; ++r) {
                int row = tileM + waveM * 32 + mi * 16 + r + rofs;
                float v = acc[mi][ni][r] + bv;
                v = v > 0.f ? v : 0.f;
                long long oi = (long long)e * strideOe + (long long)row * N + col;
                if (outB) outB[oi] = f32_to_bf16_rne(v);
                else      outF[oi] = v;
            }
        }
    }
}

// ------------------------- gate logits + softmax ----------------------------
// One wave per batch row. 20 dot products of length 1024, shfl_xor reduce,
// grouped softmax: [0,6)=task0, [6,12)=task1, [12,20)=shared.
__global__ __launch_bounds__(256)
void gates_kernel(const float* __restrict__ x, const float* __restrict__ Wg,
                  const float* __restrict__ Wgs, float* __restrict__ gbuf)
{
    const int lane = threadIdx.x & 31;
    const int wave = threadIdx.x >> 5;
    const int b = blockIdx.x * 8 + wave;
    const float* xr = x + (long long)b * D_SZ;

    float logit[NGATE];
    #pragma unroll
    for (int c = 0; c < NGATE; ++c) {
        const float* wcol;
        int ld;
        if (c < 6)       { wcol = Wg + c;                           ld = 6; }
        else if (c < 12) { wcol = Wg + (long long)D_SZ * 6 + (c-6);  ld = 6; }
        else             { wcol = Wgs + (c - 12);                    ld = 8; }
        float p = 0.f;
        for (int d = lane; d < D_SZ; d += 32)
            p += xr[d] * wcol[(long long)d * ld];
        #pragma unroll
        for (int off = 16; off; off >>= 1)
            p += __shfl_xor(p, off, 32);
        logit[c] = p;
    }

    if (lane == 0) {
        float* gout = gbuf + (long long)b * NGATE;
        int starts[3] = {0, 6, 12};
        int lens[3]   = {6, 6, 8};
        for (int grp = 0; grp < 3; ++grp) {
            int s = starts[grp], n = lens[grp];
            float m = logit[s];
            for (int i = 1; i < n; ++i) m = fmaxf(m, logit[s + i]);
            float sum = 0.f;
            for (int i = 0; i < n; ++i) {
                float t = expf(logit[s + i] - m);
                logit[s + i] = t;
                sum += t;
            }
            float inv = 1.f / sum;
            for (int i = 0; i < n; ++i) gout[s + i] = logit[s + i] * inv;
        }
    }
}

// ------------------------- gated expert combine -----------------------------
// task0 experts {0,1,4,5,6,7}; task1 {2,3,4,5,6,7}; shared all 8.
__global__ __launch_bounds__(256)
void combine_kernel(const float* __restrict__ eo, const float* __restrict__ gbuf,
                    float* __restrict__ out)
{
    long long i = (long long)blockIdx.x * 256 + threadIdx.x;  // over B*O
    int b = (int)(i / O_SZ);
    int o = (int)(i % O_SZ);
    const float* g = gbuf + (long long)b * NGATE;

    float ev[E_SZ];
    #pragma unroll
    for (int e = 0; e < E_SZ; ++e)
        ev[e] = eo[((long long)e * B_SZ + b) * O_SZ + o];

    float t0 = g[0]*ev[0] + g[1]*ev[1] + g[2]*ev[4] + g[3]*ev[5]
             + g[4]*ev[6] + g[5]*ev[7];
    float t1 = g[6]*ev[2] + g[7]*ev[3] + g[8]*ev[4] + g[9]*ev[5]
             + g[10]*ev[6] + g[11]*ev[7];
    float ts = 0.f;
    #pragma unroll
    for (int e = 0; e < E_SZ; ++e) ts += g[12 + e] * ev[e];

    long long ob = (long long)b * 3 * O_SZ + o;
    out[ob]            = t0;
    out[ob + O_SZ]     = t1;
    out[ob + 2 * O_SZ] = ts;
}

// ---------------------------------------------------------------------------
extern "C" void kernel_launch(void* const* d_in, const int* in_sizes, int n_in,
                              void* d_out, int out_size, void* d_ws, size_t ws_size,
                              hipStream_t stream) {
    (void)in_sizes; (void)n_in; (void)out_size; (void)ws_size;
    const float* x   = (const float*)d_in[0];   // [B,D]
    const float* W1  = (const float*)d_in[1];   // [E,D,H]
    const float* b1  = (const float*)d_in[2];   // [E,H]
    const float* W2  = (const float*)d_in[3];   // [E,H,O]
    const float* b2  = (const float*)d_in[4];   // [E,O]
    const float* Wg  = (const float*)d_in[5];   // [T,D,6]
    const float* Wgs = (const float*)d_in[6];   // [D,8]
    float* out = (float*)d_out;                 // [B,3,O]

    char* ws = (char*)d_ws;
    size_t off = 0;
    unsigned short* xb  = (unsigned short*)(ws + off); off += (size_t)B_SZ*D_SZ*2;       // 16 MB
    unsigned short* w1b = (unsigned short*)(ws + off); off += (size_t)E_SZ*D_SZ*H_SZ*2;  // 16 MB
    unsigned short* w2b = (unsigned short*)(ws + off); off += (size_t)E_SZ*H_SZ*O_SZ*2;  //  8 MB
    unsigned short* hb  = (unsigned short*)(ws + off); off += (size_t)E_SZ*B_SZ*H_SZ*2;  // 128 MB
    float* eo   = (float*)(ws + off);                  off += (size_t)E_SZ*B_SZ*O_SZ*4;  // 128 MB
    float* gbuf = (float*)(ws + off);                                                     // 640 KB

    // Convert inputs to bf16 (one pass; compute-bound GEMMs amortize this).
    f2bf_kernel<<<2048, 256, 0, stream>>>(x,  xb,  (long long)B_SZ * D_SZ);
    f2bf_kernel<<<2048, 256, 0, stream>>>(W1, w1b, (long long)E_SZ * D_SZ * H_SZ);
    f2bf_kernel<<<1024, 256, 0, stream>>>(W2, w2b, (long long)E_SZ * H_SZ * O_SZ);

    // Layer 1: h[e] = relu(x @ W1[e] + b1[e])   [E,B,H] (bf16)
    dim3 g1(H_SZ / 128, B_SZ / 128, E_SZ);
    gemm_bias_relu<<<g1, 256, 0, stream>>>(
        xb, 0, w1b, (long long)D_SZ * H_SZ, b1, H_SZ,
        hb, nullptr, (long long)B_SZ * H_SZ, B_SZ, H_SZ, D_SZ);

    // Layer 2: eo[e] = relu(h[e] @ W2[e] + b2[e])   [E,B,O] (f32)
    dim3 g2(O_SZ / 128, B_SZ / 128, E_SZ);
    gemm_bias_relu<<<g2, 256, 0, stream>>>(
        hb, (long long)B_SZ * H_SZ, w2b, (long long)H_SZ * O_SZ, b2, O_SZ,
        nullptr, eo, (long long)B_SZ * O_SZ, B_SZ, O_SZ, H_SZ);

    // Gates (f32 path for accuracy) then gated combine.
    gates_kernel<<<B_SZ / 8, 256, 0, stream>>>(x, Wg, Wgs, gbuf);
    combine_kernel<<<(B_SZ * O_SZ) / 256, 256, 0, stream>>>(eo, gbuf, out);
}
